// AttentionModule_25907242729618
// MI455X (gfx1250) — matrix operand order repaired and checked by reading, not compiled, not run on hardware
//
#include <hip/hip_runtime.h>

// ---------------------------------------------------------------------------
// MI455X (gfx1250) fused cross-attention module.
//  * Big GEMMs (x@Wk, x@Wv, A@Wo) on v_wmma_f32_16x16x32_bf16 (wave32 WMMA).
//  * 64-pixel tiles in kv kernel -> weight L2 re-fetch cut to ~1GB total.
//  * x pre-converted to bf16 (fused into the mean pass); tile staged to LDS
//    with global_load_async_to_lds_b128 (ASYNCcnt) via inline asm.
//  * Scores computed straight from WMMA accumulators via __shfl_xor tree
//    (K-bias dropped: it is a per-(b,n) constant, softmax-invariant).
// ---------------------------------------------------------------------------

typedef __bf16 bf16_t;
typedef __attribute__((ext_vector_type(16))) __bf16 v16bf;
typedef __attribute__((ext_vector_type(8)))  __bf16 v8bf;
typedef __attribute__((ext_vector_type(8)))  float  v8f;

#define NHEADS 8
#define CDIM   512
#define BB     16
#define PIX    (BB*64*64)   // 65536 pixels

// 16-elem bf16 fragment from two 16B-aligned 8-elem halves.
// Matches CDNA5 16-bit A layout (per-lane K-chunks at hi*8 and 16+hi*8);
// B is read with the same (p, p+16) chunks (lane = column N).
__device__ inline v16bf frag2(const bf16_t* p0, const bf16_t* p1) {
    union { v16bf v; v8bf h[2]; } u;
    u.h[0] = *(const v8bf*)p0;
    u.h[1] = *(const v8bf*)p1;
    return u.v;
}

// ---------------------------------------------------------------------------
// 1) Weights -> transposed bf16: WkT/WvT [N=512][K=512], WoT [N=512][K=64]
// ---------------------------------------------------------------------------
__global__ __launch_bounds__(256) void convert_kernel(
    const float* __restrict__ Wk, const float* __restrict__ Wv,
    const float* __restrict__ Wo,
    bf16_t* __restrict__ WkT, bf16_t* __restrict__ WvT, bf16_t* __restrict__ WoT)
{
    int i = blockIdx.x * 256 + threadIdx.x;          // 0 .. 262143
    int n = i >> 9, c = i & 511;
    WkT[i] = (bf16_t)Wk[(size_t)c * CDIM + n];
    WvT[i] = (bf16_t)Wv[(size_t)c * CDIM + n];
    if (i < 512 * 64) {
        int n2 = i >> 6, v = i & 63;
        WoT[i] = (bf16_t)Wo[(size_t)v * CDIM + n2];  // Wo is [64][512]
    }
}

// ---------------------------------------------------------------------------
// 2) Fused: x -> bf16 copy  +  partial spatial sums (16 slices per image)
// ---------------------------------------------------------------------------
__global__ __launch_bounds__(256) void meanconv_kernel(
    const float* __restrict__ x, bf16_t* __restrict__ xbf,
    float* __restrict__ part)                        // part[16][16][512]
{
    int b = blockIdx.x, cg = blockIdx.y, sl = blockIdx.z;
    int c = cg * 256 + threadIdx.x;
    const size_t base = ((size_t)b * 4096 + sl * 256) * CDIM;
    float s = 0.f;
    for (int p = 0; p < 256; ++p) {
        float v = x[base + (size_t)p * CDIM + c];
        s += v;
        xbf[base + (size_t)p * CDIM + c] = (bf16_t)v;
    }
    part[(b * 16 + sl) * CDIM + c] = s;
}

__global__ __launch_bounds__(256) void meanred_kernel(
    const float* __restrict__ part, float* __restrict__ xmean)
{
    int i = blockIdx.x * 256 + threadIdx.x;          // 16*512
    int b = i >> 9, c = i & 511;
    float s = 0.f;
    #pragma unroll
    for (int sl = 0; sl < 16; ++sl) s += part[(b * 16 + sl) * CDIM + c];
    xmean[i] = s * (1.f / 4096.f);
}

// ---------------------------------------------------------------------------
// 3) qs[b][c] = (xmean[b]@Wq + bq) * (1/sqrt(64))
// ---------------------------------------------------------------------------
__global__ __launch_bounds__(256) void q_kernel(
    const float* __restrict__ xmean, const float* __restrict__ Wq,
    const float* __restrict__ bq, float* __restrict__ qs)
{
    int b = blockIdx.x;
    for (int c = threadIdx.x; c < CDIM; c += 256) {
        float s = bq[c];
        for (int k = 0; k < CDIM; ++k)
            s += xmean[b * CDIM + k] * Wq[(size_t)k * CDIM + c];
        qs[b * CDIM + c] = s * 0.125f;
    }
}

// ---------------------------------------------------------------------------
// 4) K/V projection (WMMA). 64 pixels per block, 8 waves:
//    strip = wid&3 (16-row strip), cg = wid>>2 (256 cols = 4 heads, 16 tiles).
//    K pass -> scores via shfl_xor reduction; V pass -> bf16 V spill.
// ---------------------------------------------------------------------------
__global__ __launch_bounds__(256) void kv_kernel(
    const bf16_t* __restrict__ xbf,
    const bf16_t* __restrict__ WkT,
    const bf16_t* __restrict__ WvT, const float* __restrict__ bv,
    const float* __restrict__ qs,
    bf16_t* __restrict__ Vout, float* __restrict__ Sout)
{
    constexpr int ROWS = 64;
    constexpr int XSTR = 520;                  // 1040B rows: 16B-aligned, 4-dword bank skew
    extern __shared__ bf16_t xs[];             // 64*520*2 = 66,560 B (dynamic)

    const int tid    = threadIdx.x;
    const int pixel0 = blockIdx.x * ROWS;
    const int b      = pixel0 >> 12;

    // ---- stage x tile (bf16) into LDS via async copy (ASYNCcnt path) ----
    // VDST = per-lane LDS byte offset (low 32 bits of generic shared pointer),
    // VADDR = 64-bit global address; "off" = no SADDR.
    for (int i = tid; i < ROWS * (CDIM / 8); i += 256) {   // 4096 16B chunks
        int r = i >> 6, c8 = i & 63;
        const bf16_t* g = xbf + (size_t)(pixel0 + r) * CDIM + c8 * 8;
        unsigned ldsoff = (unsigned)(uintptr_t)(xs + r * XSTR + c8 * 8);
        asm volatile("global_load_async_to_lds_b128 %0, %1, off"
                     :: "v"(ldsoff), "v"(g) : "memory");
    }
    asm volatile("s_wait_asynccnt 0" ::: "memory");
    __syncthreads();

    const int wid  = tid >> 5;
    const int lane = tid & 31;
    const int l    = lane & 15;
    const int hi   = lane >> 4;
    const int row0 = (wid & 3) * 16;           // strip
    const int c0   = (wid >> 2) * 256;         // column group (4 heads)

    const v8f zero = {};
    v8f acc[16];

    // ---- K = x @ Wk ----
    #pragma unroll
    for (int t = 0; t < 16; ++t) acc[t] = zero;
    for (int kk = 0; kk < CDIM; kk += 32) {
        v16bf a = frag2(&xs[(row0 + l) * XSTR + kk + hi * 8],
                        &xs[(row0 + l) * XSTR + kk + hi * 8 + 16]);
        #pragma unroll
        for (int t = 0; t < 16; ++t) {
            const bf16_t* bp = &WkT[(size_t)(c0 + t * 16 + l) * CDIM + kk + hi * 8];
            acc[t] = __builtin_amdgcn_wmma_f32_16x16x32_bf16(
                false, a, false, frag2(bp, bp + 16), (short)0, acc[t], false, false);
        }
    }

    // scores: s[row][n] = sum_k K[row][n*64+k] * qs[b][n*64+k]
    // (bk omitted: per-(b,n) constant, cancelled by both softmaxes)
    {
        float qv[16];
        #pragma unroll
        for (int t = 0; t < 16; ++t) qv[t] = qs[b * CDIM + c0 + t * 16 + l];
        #pragma unroll
        for (int hloc = 0; hloc < 4; ++hloc) {
            #pragma unroll
            for (int r = 0; r < 8; ++r) {
                float p = acc[hloc * 4 + 0][r] * qv[hloc * 4 + 0]
                        + acc[hloc * 4 + 1][r] * qv[hloc * 4 + 1]
                        + acc[hloc * 4 + 2][r] * qv[hloc * 4 + 2]
                        + acc[hloc * 4 + 3][r] * qv[hloc * 4 + 3];
                p += __shfl_xor(p, 1, 32);     // reduce within 16-lane half
                p += __shfl_xor(p, 2, 32);
                p += __shfl_xor(p, 4, 32);
                p += __shfl_xor(p, 8, 32);
                if (l == 0) {
                    int row = row0 + r + hi * 8;
                    Sout[(size_t)(pixel0 + row) * NHEADS + (c0 >> 6) + hloc] = p;
                }
            }
        }
    }

    // ---- V = x @ Wv ----
    #pragma unroll
    for (int t = 0; t < 16; ++t) acc[t] = zero;
    for (int kk = 0; kk < CDIM; kk += 32) {
        v16bf a = frag2(&xs[(row0 + l) * XSTR + kk + hi * 8],
                        &xs[(row0 + l) * XSTR + kk + hi * 8 + 16]);
        #pragma unroll
        for (int t = 0; t < 16; ++t) {
            const bf16_t* bp = &WvT[(size_t)(c0 + t * 16 + l) * CDIM + kk + hi * 8];
            acc[t] = __builtin_amdgcn_wmma_f32_16x16x32_bf16(
                false, a, false, frag2(bp, bp + 16), (short)0, acc[t], false, false);
        }
    }
    #pragma unroll
    for (int t = 0; t < 16; ++t) {
        int col = c0 + t * 16 + l;
        #pragma unroll
        for (int r = 0; r < 8; ++r)
            Vout[(size_t)(pixel0 + row0 + r + hi * 8) * CDIM + col] =
                (bf16_t)(acc[t][r] + bv[col]);
    }
}

// ---------------------------------------------------------------------------
// 5) Softmax over one spatial axis + weighted sum of V.
//    mode 0: A_h[b][n][h][v] (softmax over w); mode 1: A_v[b][n][w][v].
// ---------------------------------------------------------------------------
__global__ __launch_bounds__(64) void attn_kernel(
    const float* __restrict__ S, const bf16_t* __restrict__ V,
    float* __restrict__ A, int mode)
{
    __shared__ float sh[64];
    int idx = blockIdx.x;
    int pq  = idx & 63;
    int n   = (idx >> 6) & 7;
    int b   = idx >> 9;
    int tid = threadIdx.x;

    {
        int h = mode ? tid : pq;
        int w = mode ? pq  : tid;
        sh[tid] = S[((size_t)b * 4096 + h * 64 + w) * NHEADS + n];
    }
    __syncthreads();

    float mx = -1e30f;
    for (int i = 0; i < 64; ++i) mx = fmaxf(mx, sh[i]);
    float sum = 0.f;
    for (int i = 0; i < 64; ++i) sum += __expf(sh[i] - mx);
    float rinv = 1.f / sum;

    float acc = 0.f;
    for (int i = 0; i < 64; ++i) {
        int h = mode ? i  : pq;
        int w = mode ? pq : i;
        float p = __expf(sh[i] - mx) * rinv;
        acc += p * (float)V[((size_t)b * 4096 + h * 64 + w) * CDIM + n * 64 + tid];
    }
    A[(((size_t)b * NHEADS + n) * 64 + pq) * 64 + tid] = acc;
}

// ---------------------------------------------------------------------------
// 6) out[p][:512] = (sum_n A_h[b,n,h,:] * A_v[b,n,w,:]) @ Wo + bo  (WMMA)
// ---------------------------------------------------------------------------
__global__ __launch_bounds__(256) void out_kernel(
    const float* __restrict__ Ah, const float* __restrict__ Av,
    const bf16_t* __restrict__ WoT, const float* __restrict__ bo,
    float* __restrict__ out)
{
    constexpr int ROWS = 32;
    constexpr int ASTR = 80;                   // 160B rows, 16B-aligned
    __shared__ bf16_t as[ROWS * ASTR];

    const int tid    = threadIdx.x;
    const int pixel0 = blockIdx.x * ROWS;
    const int b      = pixel0 >> 12;

    for (int i = tid; i < ROWS * 64; i += 256) {
        int r = i >> 6, v = i & 63;
        int p = pixel0 + r;
        int h = (p >> 6) & 63, w = p & 63;
        float s = 0.f;
        #pragma unroll
        for (int n = 0; n < NHEADS; ++n)
            s += Ah[(((size_t)b * NHEADS + n) * 64 + h) * 64 + v] *
                 Av[(((size_t)b * NHEADS + n) * 64 + w) * 64 + v];
        as[r * ASTR + v] = (bf16_t)s;
    }
    __syncthreads();

    const int wid = tid >> 5, lane = tid & 31, l = lane & 15, hi = lane >> 4;
    const int strip = wid & 1;
    const int cg    = wid >> 1;
    const v8f zero  = {};
    v8f acc[8];
    #pragma unroll
    for (int t = 0; t < 8; ++t) acc[t] = zero;

    #pragma unroll
    for (int kk = 0; kk < 64; kk += 32) {
        v16bf a = frag2(&as[(strip * 16 + l) * ASTR + kk + hi * 8],
                        &as[(strip * 16 + l) * ASTR + kk + hi * 8 + 16]);
        #pragma unroll
        for (int t = 0; t < 8; ++t) {
            const bf16_t* bp = &WoT[(size_t)(cg * 128 + t * 16 + l) * 64 + kk + hi * 8];
            acc[t] = __builtin_amdgcn_wmma_f32_16x16x32_bf16(
                false, a, false, frag2(bp, bp + 16), (short)0, acc[t], false, false);
        }
    }
    #pragma unroll
    for (int t = 0; t < 8; ++t) {
        int col = cg * 128 + t * 16 + l;
        #pragma unroll
        for (int r = 0; r < 8; ++r) {
            int row = strip * 16 + r + hi * 8;
            out[(size_t)(pixel0 + row) * CDIM + col] = acc[t][r] + bo[col];
        }
    }
}

// ---------------------------------------------------------------------------
extern "C" void kernel_launch(void* const* d_in, const int* in_sizes, int n_in,
                              void* d_out, int out_size, void* d_ws, size_t ws_size,
                              hipStream_t stream)
{
    (void)in_sizes; (void)n_in; (void)out_size; (void)ws_size;
    const float* x  = (const float*)d_in[0];
    const float* Wq = (const float*)d_in[1];
    const float* bq = (const float*)d_in[2];
    const float* Wk = (const float*)d_in[3];
    const float* Wv = (const float*)d_in[5];
    const float* bv = (const float*)d_in[6];
    const float* Wo = (const float*)d_in[7];
    const float* bo = (const float*)d_in[8];
    float* out = (float*)d_out;

    char* wsp = (char*)d_ws;
    auto alloc = [&](size_t bytes) -> char* {
        char* p = wsp;
        wsp += (bytes + 255) & ~(size_t)255;
        return p;
    };
    bf16_t* WkT   = (bf16_t*)alloc((size_t)CDIM * CDIM * sizeof(bf16_t)); // 512KB
    bf16_t* WvT   = (bf16_t*)alloc((size_t)CDIM * CDIM * sizeof(bf16_t)); // 512KB
    bf16_t* WoT   = (bf16_t*)alloc((size_t)CDIM * 64   * sizeof(bf16_t)); // 64KB
    float*  part  = (float*) alloc((size_t)BB * 16 * CDIM * sizeof(float));
    float*  xmean = (float*) alloc((size_t)BB * CDIM * sizeof(float));
    float*  qsb   = (float*) alloc((size_t)BB * CDIM * sizeof(float));
    float*  Sb    = (float*) alloc((size_t)PIX * NHEADS * sizeof(float)); // 2MB
    float*  Ahb   = (float*) alloc((size_t)BB * NHEADS * 64 * 64 * sizeof(float));
    float*  Avb   = (float*) alloc((size_t)BB * NHEADS * 64 * 64 * sizeof(float));
    bf16_t* Vb    = (bf16_t*)alloc((size_t)PIX * CDIM * sizeof(bf16_t));  // 64MB
    bf16_t* xbf   = (bf16_t*)alloc((size_t)PIX * CDIM * sizeof(bf16_t));  // 64MB

    const int kvLds = 64 * 520 * 2;            // 66,560 B dynamic LDS
    (void)hipFuncSetAttribute((const void*)kv_kernel,
                              hipFuncAttributeMaxDynamicSharedMemorySize, kvLds);

    convert_kernel<<<(CDIM * CDIM) / 256, 256, 0, stream>>>(Wk, Wv, Wo, WkT, WvT, WoT);
    meanconv_kernel<<<dim3(BB, 2, 16), 256, 0, stream>>>(x, xbf, part);
    meanred_kernel<<<(BB * CDIM) / 256, 256, 0, stream>>>(part, xmean);
    q_kernel<<<BB, 256, 0, stream>>>(xmean, Wq, bq, qsb);
    kv_kernel<<<PIX / 64, 256, kvLds, stream>>>(xbf, WkT, WvT, bv, qsb, Vb, Sb);
    attn_kernel<<<BB * NHEADS * 64, 64, 0, stream>>>(Sb, Vb, Ahb, 0);
    attn_kernel<<<BB * NHEADS * 64, 64, 0, stream>>>(Sb, Vb, Avb, 1);
    out_kernel<<<PIX / 32, 256, 0, stream>>>(Ahb, Avb, WoT, bo, out);
}
